// RNA_Encoder_80865644249255
// MI455X (gfx1250) — matrix operand order repaired and checked by reading, not compiled, not run on hardware
//
#include <hip/hip_runtime.h>

// ---------------------------------------------------------------------------
// Model constants (from reference)
// ---------------------------------------------------------------------------
#define BATCH      128
#define RNA_TOKENS 4000
#define RNA_COMP   400
#define EMB_RNA    10
#define EMB_DIM    256
#define N_LAYERS   6
#define D_INNER    512
#define D_STATE    16
#define D_CONV     4
#define DT_RANK    16
#define REMAIN_T   360
#define SEQ_T      361               // REMAIN_T + cls token
#define BT         (BATCH * SEQ_T)   // 46208 rows

// ---------------------------------------------------------------------------
// WMMA types (gfx1250, wave32)
// ---------------------------------------------------------------------------
typedef __attribute__((ext_vector_type(16))) __bf16 v16bf;
typedef __attribute__((ext_vector_type(8)))  float  v8f;

union FragBF {
    v16bf v;
    unsigned int w[8];
};

__device__ __forceinline__ unsigned short f2bf(float x) {
    // round-to-nearest-even f32 -> bf16
    unsigned int u = __float_as_uint(x);
    return (unsigned short)((u + 0x7FFFu + ((u >> 16) & 1u)) >> 16);
}
__device__ __forceinline__ unsigned int pack_bf(float lo, float hi) {
    return (unsigned int)f2bf(lo) | ((unsigned int)f2bf(hi) << 16);
}
__device__ __forceinline__ int imin(int a, int b) { return a < b ? a : b; }

// ---------------------------------------------------------------------------
// Generic tiled GEMM:  C[M,N] = act( A[M,K] @ B[K,N] + bias[N] )
// Block tile 128x128, 256 threads = 8 waves (4 x 2), each wave computes a
// 32x64 sub-tile = 2x4 WMMA tiles via v_wmma_f32_16x16x32_bf16.
// f32 in global; converted to packed bf16 in LDS; f32 accumulate.
// Interior blocks: software-pipelined (next tile held in registers while
// WMMAs run on the current LDS tile) + GL2 prefetch of the tile after next.
// Edge blocks: clamped loads + select.   act: 0 = identity, 1 = softplus
// ---------------------------------------------------------------------------
#define TS 34   // LDS row stride in ushorts (even -> 4B-aligned pair ops)

__global__ __launch_bounds__(256) void gemm_bf16_wmma(
    const float* __restrict__ A, const float* __restrict__ B,
    const float* __restrict__ bias, float* __restrict__ C,
    int M, int N, int K, int lda, int ldb, int ldc, int act)
{
    __shared__ unsigned short As[128 * TS];   // [m][k]   128x32 bf16
    __shared__ unsigned short Bs[128 * TS];   // [n][k]   (B tile transposed)

    const int tid  = threadIdx.x;
    const int lane = tid & 31;
    const int wave = tid >> 5;
    const int wm   = wave >> 1;     // 0..3  -> 32-row strip
    const int wn   = wave & 1;      // 0..1  -> 64-col strip
    const int hl   = lane >> 4;     // lane half (0/1)
    const int l15  = lane & 15;
    const int m0   = blockIdx.y * 128;
    const int n0   = blockIdx.x * 128;

    const bool interior = (m0 + 128 <= M) && (n0 + 128 <= N) && ((K & 31) == 0);

    v8f acc[2][4];
#pragma unroll
    for (int i = 0; i < 2; ++i)
#pragma unroll
        for (int j = 0; j < 4; ++j)
#pragma unroll
            for (int e = 0; e < 8; ++e)
                acc[i][j][e] = 0.f;

    if (interior) {
        // per-thread staging coordinates
        const int acp = tid & 15;        // A k-pair
        const int ar  = tid >> 4;        // A row base (stride 16 over j)
        const int bn  = tid & 127;       // B col
        const int bkp = tid >> 7;        // B k-pair base (stride 2 over j)

        float2 ra[8];
        float  rb0[8], rb1[8];

        // ---- prologue: load k-tile 0 into registers ----
#pragma unroll
        for (int j = 0; j < 8; ++j) {
            int r = ar + j * 16;
            ra[j] = *(const float2*)(A + (size_t)(m0 + r) * lda + 2 * acp);
        }
#pragma unroll
        for (int j = 0; j < 8; ++j) {
            int kp = bkp + j * 2;
            size_t gb = (size_t)(2 * kp) * ldb + n0 + bn;
            rb0[j] = B[gb];
            rb1[j] = B[gb + ldb];
        }

        for (int k0 = 0; k0 < K; k0 += 32) {
            __syncthreads();
            // ---- commit staged registers to LDS (packed bf16) ----
#pragma unroll
            for (int j = 0; j < 8; ++j) {
                int r = ar + j * 16;
                *(unsigned int*)&As[r * TS + 2 * acp] = pack_bf(ra[j].x, ra[j].y);
            }
#pragma unroll
            for (int j = 0; j < 8; ++j) {
                int kp = bkp + j * 2;
                *(unsigned int*)&Bs[bn * TS + 2 * kp] = pack_bf(rb0[j], rb1[j]);
            }
            __syncthreads();

            // ---- issue next tile's global loads (hidden behind WMMAs) ----
            int kn = k0 + 32;
            if (kn < K) {
#pragma unroll
                for (int j = 0; j < 8; ++j) {
                    int r = ar + j * 16;
                    ra[j] = *(const float2*)(A + (size_t)(m0 + r) * lda + kn + 2 * acp);
                }
#pragma unroll
                for (int j = 0; j < 8; ++j) {
                    int kp = bkp + j * 2;
                    size_t gb = (size_t)(kn + 2 * kp) * ldb + n0 + bn;
                    rb0[j] = B[gb];
                    rb1[j] = B[gb + ldb];
                }
                // warm GL2 for the tile after next
                if (kn + 32 < K) {
                    __builtin_prefetch(A + (size_t)(m0 + ar) * lda + kn + 32, 0, 0);
                    __builtin_prefetch(B + (size_t)(kn + 32 + 2 * bkp) * ldb + n0 + bn, 0, 0);
                }
            }

            // ---- fragments per ISA VGPR layouts ----
            FragBF fa[2], fb[4];
#pragma unroll
            for (int mt = 0; mt < 2; ++mt) {
                int row = wm * 32 + mt * 16 + l15;       // A: m = lane&15
#pragma unroll
                for (int v = 0; v < 8; ++v) {
                    // 16-bit A 16x32: k = (v>>2)*16 + half*8 + (v&3)*2
                    int k = ((v >> 2) << 4) + (hl << 3) + ((v & 3) << 1);
                    fa[mt].w[v] = *(const unsigned int*)&As[row * TS + k];
                }
            }
#pragma unroll
            for (int nt = 0; nt < 4; ++nt) {
                int col = wn * 64 + nt * 16 + l15;       // B: n = lane&15
#pragma unroll
                for (int v = 0; v < 8; ++v) {
                    // 16-bit B 32x16: k = (v>>2)*16 + half*8 + (v&3)*2
                    int k = ((v >> 2) << 4) + (hl << 3) + ((v & 3) << 1);
                    fb[nt].w[v] = *(const unsigned int*)&Bs[col * TS + k];
                }
            }
#pragma unroll
            for (int mt = 0; mt < 2; ++mt)
#pragma unroll
                for (int nt = 0; nt < 4; ++nt)
                    acc[mt][nt] = __builtin_amdgcn_wmma_f32_16x16x32_bf16(
                        false, fa[mt].v, false, fb[nt].v,
                        (short)0, acc[mt][nt], false, false);
        }

        // ---- guard-free epilogue ----
#pragma unroll
        for (int mt = 0; mt < 2; ++mt) {
#pragma unroll
            for (int nt = 0; nt < 4; ++nt) {
                int col = n0 + wn * 64 + nt * 16 + l15;
                float bv = bias ? bias[col] : 0.f;
#pragma unroll
                for (int v = 0; v < 8; ++v) {
                    int row = m0 + wm * 32 + mt * 16 + hl * 8 + v;
                    float val = acc[mt][nt][v] + bv;
                    if (act == 1)
                        val = (val > 20.f) ? val : logf(1.f + __expf(val));
                    C[(size_t)row * ldc + col] = val;
                }
            }
        }
        return;
    }

    // ======================= guarded (edge / tiny) path =====================
    for (int k0 = 0; k0 < K; k0 += 32) {
        __syncthreads();
#pragma unroll
        for (int j = 0; j < 8; ++j) {
            int i  = tid + j * 256;
            int cp = i & 15;
            int r  = i >> 4;
            int gm = m0 + r, gk = k0 + 2 * cp;
            int cm = imin(gm, M - 1);
            int c0 = imin(gk, K - 1), c1 = imin(gk + 1, K - 1);
            float a0 = A[(size_t)cm * lda + c0];
            float a1 = A[(size_t)cm * lda + c1];
            a0 = (gm < M && gk < K) ? a0 : 0.f;
            a1 = (gm < M && gk + 1 < K) ? a1 : 0.f;
            *(unsigned int*)&As[r * TS + 2 * cp] = pack_bf(a0, a1);
        }
#pragma unroll
        for (int j = 0; j < 8; ++j) {
            int i  = tid + j * 256;
            int n  = i & 127;
            int kp = i >> 7;
            int gn = n0 + n, gk = k0 + 2 * kp;
            int cn = imin(gn, N - 1);
            int c0 = imin(gk, K - 1), c1 = imin(gk + 1, K - 1);
            float b0 = B[(size_t)c0 * ldb + cn];
            float b1 = B[(size_t)c1 * ldb + cn];
            b0 = (gn < N && gk < K) ? b0 : 0.f;
            b1 = (gn < N && gk + 1 < K) ? b1 : 0.f;
            *(unsigned int*)&Bs[n * TS + 2 * kp] = pack_bf(b0, b1);
        }
        __syncthreads();

        FragBF fa[2], fb[4];
#pragma unroll
        for (int mt = 0; mt < 2; ++mt) {
            int row = wm * 32 + mt * 16 + l15;
#pragma unroll
            for (int v = 0; v < 8; ++v) {
                int k = ((v >> 2) << 4) + (hl << 3) + ((v & 3) << 1);
                fa[mt].w[v] = *(const unsigned int*)&As[row * TS + k];
            }
        }
#pragma unroll
        for (int nt = 0; nt < 4; ++nt) {
            int col = wn * 64 + nt * 16 + l15;
#pragma unroll
            for (int v = 0; v < 8; ++v) {
                int k = ((v >> 2) << 4) + (hl << 3) + ((v & 3) << 1);
                fb[nt].w[v] = *(const unsigned int*)&Bs[col * TS + k];
            }
        }
#pragma unroll
        for (int mt = 0; mt < 2; ++mt)
#pragma unroll
            for (int nt = 0; nt < 4; ++nt)
                acc[mt][nt] = __builtin_amdgcn_wmma_f32_16x16x32_bf16(
                    false, fa[mt].v, false, fb[nt].v,
                    (short)0, acc[mt][nt], false, false);
    }

#pragma unroll
    for (int mt = 0; mt < 2; ++mt) {
#pragma unroll
        for (int nt = 0; nt < 4; ++nt) {
            int col = n0 + wn * 64 + nt * 16 + l15;
            if (col >= N) continue;
            float bv = bias ? bias[col] : 0.f;
#pragma unroll
            for (int v = 0; v < 8; ++v) {
                int row = m0 + wm * 32 + mt * 16 + hl * 8 + v;
                if (row < M) {
                    float val = acc[mt][nt][v] + bv;
                    if (act == 1)
                        val = (val > 20.f) ? val : logf(1.f + __expf(val));
                    C[(size_t)row * ldc + col] = val;
                }
            }
        }
    }
}

// ---------------------------------------------------------------------------
// Gather + positional embedding + cls token:
// x0[b, t, e] = t==0 ? cls[e] : emb[b, idx, e] + pos[idx, e],  idx = fwd[t-1, b]
// ---------------------------------------------------------------------------
__global__ void gather_embed_kernel(const float* __restrict__ emb,
                                    const float* __restrict__ pos,
                                    const float* __restrict__ cls,
                                    const int*   __restrict__ fwd,
                                    float* __restrict__ x0)
{
    int bt = blockIdx.x;          // b*SEQ_T + t
    int e  = threadIdx.x;         // 0..255
    int b  = bt / SEQ_T;
    int t  = bt - b * SEQ_T;
    float v;
    if (t == 0) {
        v = cls[e];
    } else {
        int c = fwd[(t - 1) * BATCH + b];
        v = emb[((size_t)b * RNA_COMP + c) * EMB_DIM + e] + pos[(size_t)c * EMB_DIM + e];
    }
    x0[(size_t)bt * EMB_DIM + e] = v;
}

// ---------------------------------------------------------------------------
// Depthwise causal conv (width 4) + SiLU.
// u[b,t,d] = silu( sum_j xc[b,t-3+j,d]*cw[d,j] + cb[d] ),  xc = xz[:, :512]
// ---------------------------------------------------------------------------
__global__ void conv_silu_kernel(const float* __restrict__ xz,
                                 const float* __restrict__ cw,
                                 const float* __restrict__ cb,
                                 float* __restrict__ u)
{
    int i  = blockIdx.x * 256 + threadIdx.x;   // over BT*512
    int d  = i & 511;
    int bt = i >> 9;
    int b  = bt / SEQ_T;
    int t  = bt - b * SEQ_T;
    float acc = cb[d];
#pragma unroll
    for (int j = 0; j < 4; ++j) {
        int tp = t - 3 + j;
        if (tp >= 0)
            acc += xz[(size_t)(bt - 3 + j) * (2 * D_INNER) + d] * cw[d * 4 + j];
    }
    u[(size_t)bt * D_INNER + d] = acc / (1.f + __expf(-acc));
}

// ---------------------------------------------------------------------------
// Selective scan: one thread per (b, d); warp lanes share b so B/C loads are
// wave-uniform.  h[s] = exp(delta*A[d,s])*h[s] + delta*u*B[s]; y = h.C
// ---------------------------------------------------------------------------
__global__ void scan_kernel(const float* __restrict__ u,
                            const float* __restrict__ delta,
                            const float* __restrict__ xz,
                            const float* __restrict__ xdbc,
                            const float* __restrict__ A_log,
                            const float* __restrict__ Dp,
                            float* __restrict__ y)
{
    int tid = blockIdx.x * 256 + threadIdx.x;   // BATCH*D_INNER = 65536
    int b = tid >> 9;
    int d = tid & 511;

    float Aneg[D_STATE], h[D_STATE];
#pragma unroll
    for (int s = 0; s < D_STATE; ++s) {
        Aneg[s] = -__expf(A_log[d * D_STATE + s]);
        h[s]    = 0.f;
    }
    float Dd = Dp[d];

    size_t base = (size_t)b * SEQ_T;
    for (int t = 0; t < SEQ_T; ++t) {
        size_t bt = base + t;
        float del = delta[bt * D_INNER + d];
        float uu  = u[bt * D_INNER + d];
        float zz  = xz[bt * (2 * D_INNER) + D_INNER + d];
        float du  = del * uu;
        float yv  = 0.f;
        const float* Bp = xdbc + bt * (DT_RANK + 2 * D_STATE) + DT_RANK;
        const float* Cp = Bp + D_STATE;
#pragma unroll
        for (int s = 0; s < D_STATE; ++s) {
            h[s] = __expf(del * Aneg[s]) * h[s] + du * Bp[s];
            yv  += h[s] * Cp[s];
        }
        yv += uu * Dd;
        yv *= zz / (1.f + __expf(-zz));   // * silu(z)
        y[bt * D_INNER + d] = yv;
    }
}

// ---------------------------------------------------------------------------
// Final LayerNorm over E=256 and transpose [B,T,E] -> [T,B,E]
// ---------------------------------------------------------------------------
__global__ void ln_transpose_kernel(const float* __restrict__ x,
                                    const float* __restrict__ g,
                                    const float* __restrict__ beta,
                                    float* __restrict__ out)
{
    __shared__ float s1[256];
    __shared__ float s2[256];
    int bt = blockIdx.x;
    int e  = threadIdx.x;
    int b  = bt / SEQ_T;
    int t  = bt - b * SEQ_T;
    float v = x[(size_t)bt * EMB_DIM + e];
    s1[e] = v;
    s2[e] = v * v;
    __syncthreads();
    for (int s = 128; s > 0; s >>= 1) {
        if (e < s) { s1[e] += s1[e + s]; s2[e] += s2[e + s]; }
        __syncthreads();
    }
    float mu  = s1[0] * (1.f / 256.f);
    float var = s2[0] * (1.f / 256.f) - mu * mu;
    float r   = rsqrtf(var + 1e-5f);
    out[((size_t)t * BATCH + b) * EMB_DIM + e] = (v - mu) * r * g[e] + beta[e];
}

// bit-preserving pass-through of backward_indexes into output tail
__global__ void copy_idx_kernel(const int* __restrict__ src,
                                int* __restrict__ dst, int n)
{
    int i = blockIdx.x * 256 + threadIdx.x;
    if (i < n) dst[i] = src[i];
}

// ---------------------------------------------------------------------------
// Host orchestration
// ---------------------------------------------------------------------------
static void launch_gemm(hipStream_t s, const float* A, const float* B,
                        const float* bias, float* C,
                        int M, int N, int K, int lda, int ldb, int ldc, int act)
{
    dim3 grid((N + 127) / 128, (M + 127) / 128);
    gemm_bf16_wmma<<<grid, 256, 0, s>>>(A, B, bias, C, M, N, K, lda, ldb, ldc, act);
}

extern "C" void kernel_launch(void* const* d_in, const int* in_sizes, int n_in,
                              void* d_out, int out_size, void* d_ws, size_t ws_size,
                              hipStream_t stream)
{
    (void)in_sizes; (void)n_in; (void)out_size; (void)ws_size;

    const float* patches = (const float*)d_in[0];
    const int*   fwd     = (const int*)  d_in[1];
    const int*   bwd     = (const int*)  d_in[2];
    const float* W_tok   = (const float*)d_in[3];
    const float* b_tok   = (const float*)d_in[4];
    const float* W_emb   = (const float*)d_in[5];
    const float* b_emb   = (const float*)d_in[6];
    const float* cls     = (const float*)d_in[7];
    const float* pos     = (const float*)d_in[8];
    const float* ln_g    = (const float*)d_in[9];
    const float* ln_b    = (const float*)d_in[10];
    const float* W_in    = (const float*)d_in[11];
    const float* conv_w  = (const float*)d_in[12];
    const float* conv_b  = (const float*)d_in[13];
    const float* W_x     = (const float*)d_in[14];
    const float* W_dt    = (const float*)d_in[15];
    const float* b_dt    = (const float*)d_in[16];
    const float* A_log   = (const float*)d_in[17];
    const float* Dp      = (const float*)d_in[18];
    const float* W_out   = (const float*)d_in[19];

    float* ws = (float*)d_ws;
    // workspace layout (float offsets)
    const size_t OFF_TOK  = 0;                                          // 128*4000
    const size_t OFF_EMB  = OFF_TOK  + (size_t)BATCH * RNA_TOKENS;      // 51200*256
    const size_t OFF_XA   = OFF_EMB  + (size_t)BATCH * RNA_COMP * EMB_DIM;
    const size_t OFF_XB   = OFF_XA   + (size_t)BT * EMB_DIM;
    const size_t OFF_XZ   = OFF_XB   + (size_t)BT * EMB_DIM;
    const size_t OFF_U    = OFF_XZ   + (size_t)BT * 2 * D_INNER;
    const size_t OFF_XDBC = OFF_U    + (size_t)BT * D_INNER;
    const size_t OFF_DEL  = OFF_XDBC + (size_t)BT * (DT_RANK + 2 * D_STATE);
    const size_t OFF_Y    = OFF_DEL  + (size_t)BT * D_INNER;

    float* tok   = ws + OFF_TOK;
    float* emb   = ws + OFF_EMB;
    float* xA    = ws + OFF_XA;
    float* xB    = ws + OFF_XB;
    float* xz    = ws + OFF_XZ;
    float* u     = ws + OFF_U;
    float* xdbc  = ws + OFF_XDBC;
    float* delta = ws + OFF_DEL;
    float* y     = ws + OFF_Y;

    // 1) tok = patches @ W_tok + b_tok            [128,4000]x[4000,4000]
    launch_gemm(stream, patches, W_tok, b_tok, tok,
                BATCH, RNA_TOKENS, RNA_TOKENS,
                RNA_TOKENS, RNA_TOKENS, RNA_TOKENS, 0);

    // 2) emb = tok.reshape(51200,10) @ W_emb + b_emb   (K=10, guarded path)
    launch_gemm(stream, tok, W_emb, b_emb, emb,
                BATCH * RNA_COMP, EMB_DIM, EMB_RNA,
                EMB_RNA, EMB_DIM, EMB_DIM, 0);

    // 3) gather shuffled tokens + pos emb, prepend cls -> xA [B,361,256]
    gather_embed_kernel<<<BT, 256, 0, stream>>>(emb, pos, cls, fwd, xA);

    float* x_cur = xA;
    float* x_nxt = xB;

    for (int l = 0; l < N_LAYERS; ++l) {
        const float* Wi  = W_in   + (size_t)l * EMB_DIM * 2 * D_INNER;
        const float* cw  = conv_w + (size_t)l * D_INNER * D_CONV;
        const float* cb  = conv_b + (size_t)l * D_INNER;
        const float* Wx  = W_x    + (size_t)l * D_INNER * (DT_RANK + 2 * D_STATE);
        const float* Wdt = W_dt   + (size_t)l * DT_RANK * D_INNER;
        const float* bdt = b_dt   + (size_t)l * D_INNER;
        const float* Al  = A_log  + (size_t)l * D_INNER * D_STATE;
        const float* Dl  = Dp     + (size_t)l * D_INNER;
        const float* Wo  = W_out  + (size_t)l * D_INNER * EMB_DIM;

        // xz = x @ W_in                         [BT,256]x[256,1024]  (fast path)
        launch_gemm(stream, x_cur, Wi, nullptr, xz,
                    BT, 2 * D_INNER, EMB_DIM, EMB_DIM, 2 * D_INNER, 2 * D_INNER, 0);

        // u = silu(depthwise_conv(xc) + cb)
        conv_silu_kernel<<<(BT * D_INNER) / 256, 256, 0, stream>>>(xz, cw, cb, u);

        // xdbc = u @ W_x                        [BT,512]x[512,48]
        launch_gemm(stream, u, Wx, nullptr, xdbc,
                    BT, DT_RANK + 2 * D_STATE, D_INNER,
                    D_INNER, DT_RANK + 2 * D_STATE, DT_RANK + 2 * D_STATE, 0);

        // delta = softplus(xdbc[:, :16] @ W_dt + b_dt)   [BT,16]x[16,512]
        launch_gemm(stream, xdbc, Wdt, bdt, delta,
                    BT, D_INNER, DT_RANK,
                    DT_RANK + 2 * D_STATE, D_INNER, D_INNER, 1);

        // selective scan -> y (includes +u*D and *silu(z))
        scan_kernel<<<(BATCH * D_INNER) / 256, 256, 0, stream>>>(
            u, delta, xz, xdbc, Al, Dl, y);

        // x_next = y @ W_out                    [BT,512]x[512,256]  (fast path)
        launch_gemm(stream, y, Wo, nullptr, x_nxt,
                    BT, EMB_DIM, D_INNER, D_INNER, EMB_DIM, EMB_DIM, 0);

        float* tmp = x_cur; x_cur = x_nxt; x_nxt = tmp;
    }

    // LayerNorm + transpose [B,T,E]->[T,B,E] into d_out
    ln_transpose_kernel<<<BT, 256, 0, stream>>>(x_cur, ln_g, ln_b, (float*)d_out);

    // append backward_indexes (bit-preserved ints) after the features
    const int nidx = RNA_COMP * BATCH;   // 51200
    int* idx_dst = (int*)d_out + (size_t)SEQ_T * BATCH * EMB_DIM;
    copy_idx_kernel<<<(nidx + 255) / 256, 256, 0, stream>>>(bwd, idx_dst, nidx);
}
